// LSTM_medGAN_79164837200396
// MI455X (gfx1250) — compile-verified, matrix operand order checked by reading, not run on hardware
//
#include <hip/hip_runtime.h>
#include <hip/hip_bf16.h>

typedef __attribute__((ext_vector_type(4)))  float    v4f;
typedef __attribute__((ext_vector_type(4)))  _Float16 v4h;
typedef __attribute__((ext_vector_type(8)))  _Float16 v8h;
typedef __attribute__((ext_vector_type(16))) _Float16 v16h;
typedef __attribute__((ext_vector_type(8)))  float    v8f;

// ---------------------------------------------------------------------------
// WMMA helpers (CDNA5 wave32, V_WMMA_F32_16X16X32_F16)
// A fragment (16x32 f16, row-major source, stride in halves):
//   lane L<16 : row M=L,    K = {0..7, 16..23}
//   lane L>=16: row M=L-16, K = {8..15, 24..31}
// B[k][n] = W[n][k]  -> identical load pattern applied to row-major W[N][K].
// ---------------------------------------------------------------------------
__device__ __forceinline__ v16h load_frag(const _Float16* p, int stride) {
  int lane = threadIdx.x & 31;
  int r    = lane & 15;
  int kb   = (lane < 16) ? 0 : 8;
  const _Float16* q = p + r * stride + kb;
  v8h lo = *(const v8h*)(q);
  v8h hi = *(const v8h*)(q + 16);
  return __builtin_shufflevector(lo, hi, 0,1,2,3,4,5,6,7,8,9,10,11,12,13,14,15);
}

__device__ __forceinline__ v8f wmma16(v16h a, v16h b, v8f c) {
  return __builtin_amdgcn_wmma_f32_16x16x32_f16(
      /*neg_a=*/false, a, /*neg_b=*/false, b,
      /*c_mod=*/(short)0, c, /*reuse_a=*/false, /*reuse_b=*/false);
}

__device__ __forceinline__ v8f zero8() {
  v8f z = {0.f,0.f,0.f,0.f,0.f,0.f,0.f,0.f};
  return z;
}

__device__ __forceinline__ float sigmoidf(float x) { return 1.f / (1.f + expf(-x)); }

// ---------------------------------------------------------------------------
// Kernel 1: og_v[row, d] = sum_s relu(emb[seq[row,s], d]) + time_encoding
// grid = 3200 rows (b*100+t), block = 256 (= D)
// ---------------------------------------------------------------------------
__global__ void __launch_bounds__(256) embed_time_kernel(
    const int*   __restrict__ seqs,   // [3200, 48]
    const float* __restrict__ st,     // [3200]
    const float* __restrict__ emb,    // [20001, 256]
    const float* __restrict__ Wt1,    // [64]
    const float* __restrict__ bt1,    // [64]
    const float* __restrict__ Wt2,    // [256, 64]
    const float* __restrict__ bt2,    // [256]
    float*       __restrict__ ogv)    // [3200, 256]
{
  __shared__ float tf[64];
  int row = blockIdx.x;
  int d   = threadIdx.x;
  float tt = st[row] * (1.f / 180.f);
  if (d < 64) {
    float x = tt * Wt1[d] + bt1[d];
    tf[d] = 1.f - tanhf(x * x);
  }
  __syncthreads();
  float acc = 0.f;
  const int* sp = seqs + row * 48;
  #pragma unroll 4
  for (int s = 0; s < 48; ++s) {
    int id  = sp[s];
    float e = emb[id * 256 + d];
    acc += fmaxf(e, 0.f);
  }
  float te = bt2[d];
  #pragma unroll 8
  for (int j = 0; j < 64; ++j) te += tf[j] * Wt2[d * 64 + j];
  ogv[row * 256 + d] = acc + te;
}

// ---------------------------------------------------------------------------
// Kernel 2: generic WMMA GEMM  Y[M,N] = act(X[M,K] @ W[N,K]^T + b0 + b1)
// f32 in/out, f16 compute. Block = 256 thr = 8 waves, block tile 32x64,
// K staged in chunks of 64 (2 WMMAs per wave per stage).
// Staging is float4 -> packed-cvt -> v4h (b128 global loads, b64 LDS stores).
// Requires M%32==0, N%64==0, K%64==0, 16B-aligned operands.
// ---------------------------------------------------------------------------
#define BM 32
#define BN 64
#define BK 64

__global__ void __launch_bounds__(256) gemm_f16_kernel(
    const float* __restrict__ X, int ldx,
    const float* __restrict__ W,           // [N, K] row-major
    const float* __restrict__ b0,
    const float* __restrict__ b1,
    float*       __restrict__ Y, int ldy,
    int M, int N, int K, int relu)
{
  __shared__ __align__(16) _Float16 xs[BM * BK];   // 4 KB
  __shared__ __align__(16) _Float16 wsm[BN * BK];  // 8 KB
  int tid  = threadIdx.x;
  int wave = tid >> 5;
  int m0   = blockIdx.y * BM;
  int n0   = blockIdx.x * BN;
  int wm   = wave >> 2;     // 0..1  (M sub-tile)
  int wn   = wave & 3;      // 0..3  (N sub-tile)

  v8f acc = zero8();

  for (int k0 = 0; k0 < K; k0 += BK) {
    // ---- stage X tile (32 x 64): 512 float4 chunks, 2 per thread ----
    #pragma unroll
    for (int it = 0; it < 2; ++it) {
      int q = tid + it * 256;
      int r = q >> 4;              // 0..31
      int c = (q & 15) * 4;        // 0..60
      v4f f = *(const v4f*)(X + (m0 + r) * ldx + k0 + c);
      *(v4h*)(xs + r * BK + c) = __builtin_convertvector(f, v4h);
    }
    // ---- stage W tile (64 x 64): 1024 float4 chunks, 4 per thread ----
    #pragma unroll
    for (int it = 0; it < 4; ++it) {
      int q = tid + it * 256;
      int r = q >> 4;              // 0..63
      int c = (q & 15) * 4;
      v4f f = *(const v4f*)(W + (n0 + r) * K + k0 + c);
      *(v4h*)(wsm + r * BK + c) = __builtin_convertvector(f, v4h);
    }
    if (k0 + BK < K) {
      __builtin_prefetch(X + (m0 + (tid >> 4)) * ldx + k0 + BK, 0, 1);
      __builtin_prefetch(W + (n0 + (tid >> 2)) * K + k0 + BK, 0, 1);
    }
    __syncthreads();
    #pragma unroll
    for (int kk = 0; kk < BK; kk += 32) {
      v16h a = load_frag(xs  + (wm * 16) * BK + kk, BK);
      v16h b = load_frag(wsm + (wn * 16) * BK + kk, BK);
      acc = wmma16(a, b, acc);
    }
    __syncthreads();
  }

  int lane = tid & 31;
  int n    = n0 + wn * 16 + (lane & 15);
  int mb   = m0 + wm * 16 + ((lane < 16) ? 0 : 8);
  float bv = (b0 ? b0[n] : 0.f) + (b1 ? b1[n] : 0.f);
  #pragma unroll
  for (int r = 0; r < 8; ++r) {
    float v = acc[r] + bv;
    if (relu) v = fmaxf(v, 0.f);
    Y[(mb + r) * ldy + n] = v;
  }
}

// ---------------------------------------------------------------------------
// Kernel 3: f32 -> f16 convert (for Whh)
// ---------------------------------------------------------------------------
__global__ void f32_to_f16_kernel(const float* __restrict__ X,
                                  _Float16* __restrict__ Y, int n) {
  int i = (blockIdx.x * blockDim.x + threadIdx.x) * 4;
  if (i < n) {
    v4f f = *(const v4f*)(X + i);
    *(v4h*)(Y + i) = __builtin_convertvector(f, v4h);
  }
}

// ---------------------------------------------------------------------------
// Kernel 4: LSTM recurrence, single persistent workgroup, 512 thr = 16 waves.
// h state in LDS (32x256 f16), c state in registers, Whh f16 streamed from L2.
// Wave w owns hidden units [16w, 16w+16) for ALL four gates and both batch
// halves -> gate fusion is lane-local (no cross-wave exchange).
// 64 wmma / wave / step (1024 per step), 100 steps.
// ---------------------------------------------------------------------------
__global__ void __launch_bounds__(512) lstm_rec_kernel(
    const float*    __restrict__ xg,    // [3200, 1024], row = b*100+t, gates [i f g o]
    const _Float16* __restrict__ whh,   // [1024, 256] f16
    float*          __restrict__ og_h,  // [3200, 256] (d_out slice)
    float*          __restrict__ h_n)   // [32, 256]   (d_out slice)
{
  __shared__ __align__(16) _Float16 hlds[32 * 256];   // 16 KB
  int tid  = threadIdx.x;
  int wave = tid >> 5;                 // 0..15 == u (hidden tile)
  int lane = tid & 31;
  int r15  = lane & 15;
  int mb   = (lane < 16) ? 0 : 8;
  int u    = wave;
  int j    = u * 16 + r15;             // hidden column this lane owns (C-layout N)

  for (int i = tid; i < 32 * 256; i += 512) hlds[i] = (_Float16)0.f;
  float c[2][8];
  #pragma unroll
  for (int m = 0; m < 2; ++m)
    #pragma unroll
    for (int r = 0; r < 8; ++r) c[m][r] = 0.f;
  __syncthreads();

  for (int t = 0; t < 100; ++t) {
    v8f acc[4][2];
    #pragma unroll
    for (int g = 0; g < 4; ++g) { acc[g][0] = zero8(); acc[g][1] = zero8(); }

    // h @ Whh^T  over K=256 in 8 chunks of 32
    #pragma unroll 2
    for (int kc = 0; kc < 8; ++kc) {
      int k0 = kc * 32;
      v16h a0 = load_frag(hlds + 0 * 256 + k0, 256);   // batch rows 0..15
      v16h a1 = load_frag(hlds + 16 * 256 + k0, 256);  // batch rows 16..31
      #pragma unroll
      for (int g = 0; g < 4; ++g) {
        v16h b = load_frag(whh + (g * 256 + u * 16) * 256 + k0, 256);
        acc[g][0] = wmma16(a0, b, acc[g][0]);
        acc[g][1] = wmma16(a1, b, acc[g][1]);
      }
    }

    // pointwise gate fusion; buffer new h, then publish to LDS
    float hbuf[2][8];
    #pragma unroll
    for (int m = 0; m < 2; ++m) {
      #pragma unroll
      for (int r = 0; r < 8; ++r) {
        int b   = m * 16 + mb + r;
        int row = b * 100 + t;
        const float* xr = xg + row * 1024;
        float gi = acc[0][m][r] + xr[0   + j];
        float gf = acc[1][m][r] + xr[256 + j];
        float gg = acc[2][m][r] + xr[512 + j];
        float go = acc[3][m][r] + xr[768 + j];
        float cn = sigmoidf(gf) * c[m][r] + sigmoidf(gi) * tanhf(gg);
        c[m][r]  = cn;
        float h  = sigmoidf(go) * tanhf(cn);
        hbuf[m][r] = h;
        og_h[row * 256 + j] = h;
        if (t == 99) h_n[b * 256 + j] = h;
      }
    }
    __syncthreads();   // all A-frag reads of old h done
    #pragma unroll
    for (int m = 0; m < 2; ++m)
      #pragma unroll
      for (int r = 0; r < 8; ++r)
        hlds[(m * 16 + mb + r) * 256 + j] = (_Float16)hbuf[m][r];
    __syncthreads();   // new h visible before next step
  }
}

// ---------------------------------------------------------------------------
// Kernel 5: BatchNorm(mean/var over rows, biased) + ReLU. grid = C channels.
// ---------------------------------------------------------------------------
__global__ void __launch_bounds__(256) bn_relu_kernel(
    const float* __restrict__ X, float* __restrict__ Y,
    const float* __restrict__ gamma, const float* __restrict__ beta,
    int rows, int C)
{
  __shared__ float s1[256], s2[256];
  int cch = blockIdx.x;
  int tid = threadIdx.x;
  float sum = 0.f, sq = 0.f;
  for (int r = tid; r < rows; r += 256) {
    float v = X[r * C + cch];
    sum += v; sq += v * v;
  }
  s1[tid] = sum; s2[tid] = sq;
  __syncthreads();
  for (int off = 128; off > 0; off >>= 1) {
    if (tid < off) { s1[tid] += s1[tid + off]; s2[tid] += s2[tid + off]; }
    __syncthreads();
  }
  float mu  = s1[0] / rows;
  float var = s2[0] / rows - mu * mu;
  float inv = rsqrtf(var + 1e-5f);
  float g = gamma[cch], b = beta[cch];
  for (int r = tid; r < rows; r += 256) {
    float v = X[r * C + cch];
    v = g * (v - mu) * inv + b;
    Y[r * C + cch] = fmaxf(v, 0.f);
  }
}

// ---------------------------------------------------------------------------
// Kernel 6: final classifier layer (N=2) + output assembly of f, final_fake.
// ---------------------------------------------------------------------------
__global__ void final_kernel(const float* __restrict__ h2og,
                             const float* __restrict__ h2fk,
                             const float* __restrict__ c3W,  // [2,512]
                             const float* __restrict__ c3b,  // [2]
                             float* __restrict__ out)        // d_out base
{
  int tid = threadIdx.x;
  if (tid >= 64) return;
  int b = tid >> 1, k = tid & 1;
  float so = c3b[k], sf = c3b[k];
  const float* wr = c3W + k * 512;
  for (int i = 0; i < 512; ++i) {
    float w = wr[i];
    so += w * h2og[b * 512 + i];
    sf += w * h2fk[b * 512 + i];
  }
  out[b * 2 + k]      = so + sf;  // f = final_og + final_fake
  out[64 + b * 2 + k] = sf;       // final_fake
}

// ---------------------------------------------------------------------------
// Host launch
// ---------------------------------------------------------------------------
static inline void gemm(hipStream_t s, const float* X, int ldx, const float* W,
                        const float* b0, const float* b1, float* Y, int ldy,
                        int M, int N, int K, int relu) {
  dim3 grid(N / 64, M / 32);
  gemm_f16_kernel<<<grid, 256, 0, s>>>(X, ldx, W, b0, b1, Y, ldy, M, N, K, relu);
}

extern "C" void kernel_launch(void* const* d_in, const int* in_sizes, int n_in,
                              void* d_out, int out_size, void* d_ws, size_t ws_size,
                              hipStream_t stream) {
  (void)in_sizes; (void)n_in; (void)out_size; (void)ws_size;

  const int*   seqs  = (const int*)  d_in[0];
  const float* st    = (const float*)d_in[1];
  const float* z     = (const float*)d_in[2];
  const float* emb   = (const float*)d_in[3];
  const float* Wt1   = (const float*)d_in[4];
  const float* bt1   = (const float*)d_in[5];
  const float* Wt2   = (const float*)d_in[6];
  const float* bt2   = (const float*)d_in[7];
  const float* Wih   = (const float*)d_in[8];
  const float* Whh   = (const float*)d_in[9];
  const float* bih   = (const float*)d_in[10];
  const float* bhh   = (const float*)d_in[11];
  const float* encW  = (const float*)d_in[12];
  const float* encB  = (const float*)d_in[13];
  const float* encG  = (const float*)d_in[14];
  const float* encBt = (const float*)d_in[15];
  const float* decW  = (const float*)d_in[16];
  const float* decB  = (const float*)d_in[17];
  const float* decG  = (const float*)d_in[18];
  const float* decBt = (const float*)d_in[19];
  const float* genW  = (const float*)d_in[20];
  const float* genB  = (const float*)d_in[21];
  const float* c1W   = (const float*)d_in[22];
  const float* c1B   = (const float*)d_in[23];
  const float* c2W   = (const float*)d_in[24];
  const float* c2B   = (const float*)d_in[25];
  const float* c3W   = (const float*)d_in[26];
  const float* c3B   = (const float*)d_in[27];

  // workspace layout (floats)  — ~25.5 MB total; all offsets 16B-aligned
  float* fws  = (float*)d_ws;
  float* ogv  = fws;               // 3200*256
  float* xg   = ogv  + 819200;     // 3200*1024
  float* encx = xg   + 3276800;    // 3200*64
  float* gz   = encx + 204800;     // 3200*64
  float* dec1 = gz   + 204800;     // 3200*256
  float* dec2 = dec1 + 819200;     // 3200*256
  float* h1og = dec2 + 819200;     // 32*1024
  float* h1fk = h1og + 32768;      // 32*1024
  float* h2og = h1fk + 32768;      // 32*512
  float* h2fk = h2og + 16384;      // 32*512
  _Float16* whh16 = (_Float16*)(h2fk + 16384);  // 1024*256 halves

  // output layout (floats)
  float* out       = (float*)d_out;
  float* out_hn    = out + 128;       // 32*256
  float* out_DecX  = out + 8320;      // 3200*256
  float* out_DecGz = out + 827520;    // 3200*256
  float* out_ogh   = out + 1646720;   // 3200*256

  // 0) Whh -> f16 (L2-resident operand for the recurrence)
  f32_to_f16_kernel<<<256, 256, 0, stream>>>(Whh, whh16, 262144);

  // 1) embedding gather + time encoding -> og_v
  embed_time_kernel<<<3200, 256, 0, stream>>>(seqs, st, emb, Wt1, bt1, Wt2, bt2, ogv);

  // 2) xg = og_v @ Wih^T + bih + bhh   (3200 x 1024, K=256)
  gemm(stream, ogv, 256, Wih, bih, bhh, xg, 1024, 3200, 1024, 256, 0);

  // 3) LSTM recurrence -> og_h (d_out), h_n (d_out)
  lstm_rec_kernel<<<1, 512, 0, stream>>>(xg, whh16, out_ogh, out_hn);

  // 4) EncX = bn_relu(og_h @ enc_W^T + enc_b)
  gemm(stream, out_ogh, 256, encW, encB, nullptr, encx, 64, 3200, 64, 256, 0);
  bn_relu_kernel<<<64, 256, 0, stream>>>(encx, encx, encG, encBt, 3200, 64);

  // 5) Gz = z @ gen_W^T + gen_b
  gemm(stream, z, 256, genW, genB, nullptr, gz, 64, 3200, 64, 256, 0);

  // 6) DecX = bn_relu(EncX @ dec_W^T + dec_b); DecGz = bn_relu(Gz @ dec_W^T + dec_b)
  gemm(stream, encx, 64, decW, decB, nullptr, dec1, 256, 3200, 256, 64, 0);
  gemm(stream, gz,   64, decW, decB, nullptr, dec2, 256, 3200, 256, 64, 0);
  bn_relu_kernel<<<256, 256, 0, stream>>>(dec1, out_DecX,  decG, decBt, 3200, 256);
  bn_relu_kernel<<<256, 256, 0, stream>>>(dec2, out_DecGz, decG, decBt, 3200, 256);

  // 7) classifier on last timestep only (h_n == og_h[:,-1,:]; DecGz[:,-1,:] via row stride)
  gemm(stream, out_hn, 256, c1W, c1B, nullptr, h1og, 1024, 32, 1024, 256, 1);
  gemm(stream, out_DecGz + 99 * 256, 25600, c1W, c1B, nullptr, h1fk, 1024, 32, 1024, 256, 1);
  gemm(stream, h1og, 1024, c2W, c2B, nullptr, h2og, 512, 32, 512, 1024, 1);
  gemm(stream, h1fk, 1024, c2W, c2B, nullptr, h2fk, 512, 32, 512, 1024, 1);
  final_kernel<<<1, 64, 0, stream>>>(h2og, h2fk, c3W, c3B, out);
}
